// AttnReadout_26096221290897
// MI455X (gfx1250) — compile-verified, matrix operand order checked by reading, not run on hardware
//
#include <hip/hip_runtime.h>
#include <hip/hip_bf16.h>
#include <math.h>

// ---------------------------------------------------------------------------
// MI455X (gfx1250) implementation of the hybrid attention readout.
//   * bf16 WMMA (v_wmma_f32_16x16x32_bf16) for the 1M x 128 x 128 projection
//   * native bf16 conversion (v_cvt_pk_bf16_f32)
//   * non-temporal loads for the 2x512MB feature streams (keep L2 for the
//     32KB Wu fragments / 8MB gates / 8MB e-arrays)
//   * bias bu folded into the per-graph gate vectors -> WMMA C starts at
//     inline 0 (no accumulator splat movs)
//   * contiguous-segment softmax (seg_ids sorted; last_nodes = prefix ends)
// ---------------------------------------------------------------------------

typedef __attribute__((ext_vector_type(16))) __bf16 v16bf;
typedef __attribute__((ext_vector_type(8)))  float  v8f;
typedef __attribute__((ext_vector_type(4)))  float  fx4;   // clang vector: OK for NT builtins

#define DD 128   // feature / hidden dim

__device__ __forceinline__ __bf16 f2bf(float f) {
  return (__bf16)f;                       // hardware cvt on gfx1250
}

__device__ __forceinline__ float sigmoidf_(float x) {
  return 1.0f / (1.0f + __expf(-x));
}

// ---------------------------------------------------------------------------
// Kernel 1: Wu [128,128] f32 -> bf16 B-fragments, pre-swizzled to the CDNA5
// 16-bit B-matrix VGPR layout. frag[kt][nt][lane][j], j=0..15 bf16 per lane.
//   lane<16 : K = {0..7, 16..23}  (of the 32-wide K chunk)
//   lane>=16: K = {8..15, 24..31}
//   column  h = nt*16 + (lane & 15), B[k][n] = Wu[h][d]  (B = Wu^T chunk)
// ---------------------------------------------------------------------------
__global__ void __launch_bounds__(256) build_wu_frags(
    const float* __restrict__ Wu, unsigned short* __restrict__ frag) {
  int t = blockIdx.x * blockDim.x + threadIdx.x;      // 0 .. 4*8*32*16-1
  if (t >= 4 * 8 * 32 * 16) return;
  int j    = t & 15;
  int lane = (t >> 4) & 31;
  int nt   = (t >> 9) & 7;
  int kt   = t >> 12;
  int h     = nt * 16 + (lane & 15);
  int kbase = (lane < 16) ? 0 : 8;
  int k     = (j < 8) ? (kbase + j) : (kbase + 8 + j);
  int d     = kt * 32 + k;
  union { __bf16 b; unsigned short s; } u;
  u.b = f2bf(Wu[h * DD + d]);
  frag[t] = u.s;
}

// ---------------------------------------------------------------------------
// Kernel 2: gate vectors with bias folded in:
//   vi[g] = Wv @ x_invar[last[g]] + bu,   vv[g] = Wv @ x_var[last[g]] + bu
// One 128-thread block per graph; the gathered rows are staged in LDS.
// ---------------------------------------------------------------------------
__global__ void __launch_bounds__(128) attn_gates(
    const float* __restrict__ x_invar, const float* __restrict__ x_var,
    const float* __restrict__ Wv, const float* __restrict__ bu,
    const int* __restrict__ last_nodes,
    float* __restrict__ vi, float* __restrict__ vv) {
  const int g = blockIdx.x;
  const int h = threadIdx.x;
  __shared__ float xi[DD];
  __shared__ float xv[DD];
  const int ln = last_nodes[g];
  xi[h] = x_invar[(size_t)ln * DD + h];
  xv[h] = x_var[(size_t)ln * DD + h];
  __syncthreads();
  const float bias = bu[h];
  float ai = bias, av = bias;
  const float* w = Wv + (size_t)h * DD;
#pragma unroll 8
  for (int d = 0; d < DD; ++d) {
    const float wd = w[d];
    ai += wd * xi[d];
    av += wd * xv[d];
  }
  vi[(size_t)g * DD + h] = ai;
  vv[(size_t)g * DD + h] = av;
}

// ---------------------------------------------------------------------------
// Kernel 3: main WMMA pass over 2N virtual rows (rows [0,N) = x_invar,
// rows [N,2N) = x_var). One wave = one 16-row tile:
//   p(16x128) = A(16x128 bf16) x Wu^T(128x128 bf16)          (32 WMMAs, C=0)
//   e_i[r] = sum_h sigmoid(p + vi[seg]) * We[h]   (and e_v with vv;
//            vi/vv already contain bu)
// ---------------------------------------------------------------------------
__global__ void __launch_bounds__(256) attn_pass1(
    const float* __restrict__ x_invar, const float* __restrict__ x_var,
    const unsigned short* __restrict__ wuFrag,
    const float* __restrict__ vi, const float* __restrict__ vv,
    const float* __restrict__ We, const int* __restrict__ seg_ids,
    float* __restrict__ e_i, float* __restrict__ e_v,
    int n_nodes, int two_n) {
  const int lane = threadIdx.x & 31;
  const int wave = threadIdx.x >> 5;
  const int tile = blockIdx.x * 8 + wave;
  const int base = tile * 16;
  if (base >= two_n) return;                 // wave-uniform: EXEC stays all-1s

  // ---- A fragments: 16 rows x (4 x K=32) bf16, ISA 16-bit A layout ----
  int rA = base + (lane & 15);
  if (rA > two_n - 1) rA = two_n - 1;        // clamp tail (dupes never written)
  const float* src = (rA < n_nodes) ? (x_invar + (size_t)rA * DD)
                                    : (x_var + (size_t)(rA - n_nodes) * DD);
  const int khalf = (lane >> 4) ? 8 : 0;
  v16bf aF[4];
#pragma unroll
  for (int kt = 0; kt < 4; ++kt) {
    const int d0 = kt * 32 + khalf;
    // streaming data: non-temporal so Wu frags / gates / e stay in L2
    const fx4 q0 = __builtin_nontemporal_load((const fx4*)(src + d0));
    const fx4 q1 = __builtin_nontemporal_load((const fx4*)(src + d0 + 4));
    const fx4 q2 = __builtin_nontemporal_load((const fx4*)(src + d0 + 16));
    const fx4 q3 = __builtin_nontemporal_load((const fx4*)(src + d0 + 20));
    const float tmp[16] = {q0[0], q0[1], q0[2], q0[3], q1[0], q1[1], q1[2], q1[3],
                           q2[0], q2[1], q2[2], q2[3], q3[0], q3[1], q3[2], q3[3]};
#pragma unroll
    for (int j = 0; j < 16; ++j) aF[kt][j] = f2bf(tmp[j]);
  }

  // rows owned by this lane's C registers: M = i + 8*(lane>=16)
  const int hgrp = lane >> 4;
  int segr[8];
#pragma unroll
  for (int i = 0; i < 8; ++i) {
    int r = base + i + 8 * hgrp;
    if (r > two_n - 1) r = two_n - 1;
    const int n = (r < n_nodes) ? r : r - n_nodes;
    segr[i] = seg_ids[n];
  }

  float psi[8], psv[8];
#pragma unroll
  for (int i = 0; i < 8; ++i) { psi[i] = 0.f; psv[i] = 0.f; }

  const int hcol = lane & 15;
#pragma unroll
  for (int nt = 0; nt < 8; ++nt) {
    const int h = nt * 16 + hcol;            // all 8 C VGPRs share column h
    v8f c = {};                              // first WMMA uses inline-0 SRC2
#pragma unroll
    for (int kt = 0; kt < 4; ++kt) {
      const v16bf bF =
          *(const v16bf*)(const void*)(wuFrag + (((kt * 8 + nt) * 32 + lane) << 4));
      c = __builtin_amdgcn_wmma_f32_16x16x32_bf16(
              false, aF[kt], false, bF, (short)0, c, false, false);
    }
    const float weh = We[h];
#pragma unroll
    for (int i = 0; i < 8; ++i) {
      const float fu = c[i];
      const float gi = vi[(size_t)segr[i] * DD + h];   // includes bu[h]
      const float gv = vv[(size_t)segr[i] * DD + h];   // includes bu[h]
      psi[i] += sigmoidf_(fu + gi) * weh;
      psv[i] += sigmoidf_(fu + gv) * weh;
    }
  }

  // reduce over the 16 columns held by each lane half
#pragma unroll
  for (int i = 0; i < 8; ++i) {
#pragma unroll
    for (int m = 1; m < 16; m <<= 1) {
      psi[i] += __shfl_xor(psi[i], m, 32);
      psv[i] += __shfl_xor(psv[i], m, 32);
    }
  }
  if (hcol == 0) {
#pragma unroll
    for (int i = 0; i < 8; ++i) {
      const int r = base + i + 8 * hgrp;
      if (r < two_n) { e_i[r] = psi[i]; e_v[r] = psv[i]; }
    }
  }
}

// ---------------------------------------------------------------------------
// Kernel 4: per-graph softmax + weighted sums. Segments are contiguous:
// graph g owns nodes [s, e) and virtual rows {[s,e)} U {N+[s,e)}.
// ---------------------------------------------------------------------------
__global__ void __launch_bounds__(128) attn_finalize(
    const float* __restrict__ x_invar, const float* __restrict__ x_var,
    const float* __restrict__ e_i, const float* __restrict__ e_v,
    const int* __restrict__ last_nodes, float* __restrict__ out,
    int n_nodes, int n_graphs) {
  const int g = blockIdx.x;
  const int t = threadIdx.x;
  const int s  = (g == 0) ? 0 : (last_nodes[g - 1] + 1);
  const int en = last_nodes[g] + 1;
  __shared__ float red[DD];

  float mi = -3.402823466e38f, mv = -3.402823466e38f;
  for (int n = s + t; n < en; n += DD) {
    mi = fmaxf(mi, fmaxf(e_i[n], e_i[n_nodes + n]));
    mv = fmaxf(mv, fmaxf(e_v[n], e_v[n_nodes + n]));
  }
  red[t] = mi; __syncthreads();
  for (int o = 64; o > 0; o >>= 1) { if (t < o) red[t] = fmaxf(red[t], red[t + o]); __syncthreads(); }
  mi = red[0]; __syncthreads();
  red[t] = mv; __syncthreads();
  for (int o = 64; o > 0; o >>= 1) { if (t < o) red[t] = fmaxf(red[t], red[t + o]); __syncthreads(); }
  mv = red[0]; __syncthreads();

  float di = 0.f, dv = 0.f;
  for (int n = s + t; n < en; n += DD) {
    di += __expf(e_i[n] - mi) + __expf(e_i[n_nodes + n] - mi);
    dv += __expf(e_v[n] - mv) + __expf(e_v[n_nodes + n] - mv);
  }
  red[t] = di; __syncthreads();
  for (int o = 64; o > 0; o >>= 1) { if (t < o) red[t] += red[t + o]; __syncthreads(); }
  di = red[0]; __syncthreads();
  red[t] = dv; __syncthreads();
  for (int o = 64; o > 0; o >>= 1) { if (t < o) red[t] += red[t + o]; __syncthreads(); }
  dv = red[0]; __syncthreads();

  const float ri = 1.0f / di, rv = 1.0f / dv;
  float acc_i = 0.f, acc_v = 0.f;
  for (int n = s; n < en; ++n) {
    const float wi1 = __expf(e_i[n] - mi) * ri;
    const float wi2 = __expf(e_i[n_nodes + n] - mi) * ri;
    const float wv1 = __expf(e_v[n] - mv) * rv;
    const float wv2 = __expf(e_v[n_nodes + n] - mv) * rv;
    const float a = __builtin_nontemporal_load(&x_invar[(size_t)n * DD + t]);
    const float b = __builtin_nontemporal_load(&x_var[(size_t)n * DD + t]);
    acc_i += a * wi1 + b * wi2;
    acc_v += a * wv1 + b * wv2;
  }
  out[(size_t)g * DD + t]              = acc_i;   // rst_invar [B,1,128]
  out[(size_t)(n_graphs + g) * DD + t] = acc_v;   // rst_var   [B,1,128]
}

// ---------------------------------------------------------------------------
extern "C" void kernel_launch(void* const* d_in, const int* in_sizes, int n_in,
                              void* d_out, int out_size, void* d_ws, size_t ws_size,
                              hipStream_t stream) {
  const float* x_invar = (const float*)d_in[0];
  const float* x_var   = (const float*)d_in[1];
  const float* Wu      = (const float*)d_in[2];
  const float* bu      = (const float*)d_in[3];
  const float* Wv      = (const float*)d_in[4];
  const float* We      = (const float*)d_in[5];
  const int*   seg_ids = (const int*)d_in[6];
  const int*   last    = (const int*)d_in[7];

  const int n_nodes  = in_sizes[0] / DD;   // N
  const int n_graphs = in_sizes[7];        // B
  const int two_n    = 2 * n_nodes;

  // workspace carve-up (~16.1 MB total)
  char*  ws  = (char*)d_ws;
  size_t off = 0;
  unsigned short* wuFrag = (unsigned short*)(ws + off);
  off += (size_t)4 * 8 * 32 * 16 * sizeof(unsigned short);        // 32 KB
  off = (off + 511) & ~(size_t)511;
  float* vi = (float*)(ws + off); off += (size_t)n_graphs * DD * sizeof(float);
  off = (off + 511) & ~(size_t)511;
  float* vv = (float*)(ws + off); off += (size_t)n_graphs * DD * sizeof(float);
  off = (off + 511) & ~(size_t)511;
  float* e_i = (float*)(ws + off); off += (size_t)two_n * sizeof(float);
  off = (off + 511) & ~(size_t)511;
  float* e_v = (float*)(ws + off);

  build_wu_frags<<<(4 * 8 * 32 * 16 + 255) / 256, 256, 0, stream>>>(Wu, wuFrag);
  attn_gates<<<n_graphs, DD, 0, stream>>>(x_invar, x_var, Wv, bu, last, vi, vv);

  const int tiles = (two_n + 15) / 16;
  attn_pass1<<<(tiles + 7) / 8, 256, 0, stream>>>(
      x_invar, x_var, wuFrag, vi, vv, We, seg_ids, e_i, e_v, n_nodes, two_n);

  attn_finalize<<<n_graphs, DD, 0, stream>>>(
      x_invar, x_var, e_i, e_v, last, (float*)d_out, n_nodes, n_graphs);
}
